// GroupedQueryAttn_42606075576438
// MI455X (gfx1250) — compile-verified, matrix operand order checked by reading, not run on hardware
//
#include <hip/hip_runtime.h>
#include <hip/hip_bf16.h>

// ---------------------------------------------------------------------------
// GQA attention block for MI455X (gfx1250, wave32, WMMA bf16 + async-LDS)
// ---------------------------------------------------------------------------

#define MODEL_DIM 1024
#define SEQ 2048
#define BATCH 2
#define NQH 16
#define NG 4
#define HD 64
#define QKV_OUT 1536
#define ROWS (BATCH * SEQ)          // 4096

#define MQ 128                      // queries per block in flash attention
#define KPAD 72                     // K tile row stride (64 + 8) elems
#define VPAD 40                     // V tile row stride (32 + 8) elems

typedef __bf16 bf16;
typedef __attribute__((ext_vector_type(16))) __bf16 v16bf;
typedef __attribute__((ext_vector_type(8)))  __bf16 bf16x8;
typedef __attribute__((ext_vector_type(8)))  float  v8f;
typedef __attribute__((ext_vector_type(4)))  int    vi4;

// Address-space qualified pointer types matching the async-LDS builtin
// prototype: (v4i addrspace(1)*, v4i addrspace(3)*, imm, imm).
typedef __attribute__((address_space(1))) vi4 as1_vi4;
typedef __attribute__((address_space(3))) vi4 as3_vi4;

#if defined(__has_builtin)
#if __has_builtin(__builtin_amdgcn_global_load_async_to_lds_b128) && \
    __has_builtin(__builtin_amdgcn_s_wait_asynccnt)
#define HAVE_ASYNC_LDS 1
#endif
#endif
#ifndef HAVE_ASYNC_LDS
#define HAVE_ASYNC_LDS 0
#endif

// D = A(16x32 bf16) * B(32x16 bf16) + C (f32), one matrix per wave32
__device__ __forceinline__ v8f wmma_bf16(v16bf a, v16bf b, v8f c) {
    return __builtin_amdgcn_wmma_f32_16x16x32_bf16(
        /*neg_a=*/false, a, /*neg_b=*/false, b,
        /*c_mod=*/(short)0, c, /*reuse_a=*/false, /*reuse_b=*/false);
}

// Load a 16x32 fragment (A layout; also valid for B when the source is the
// K-contiguous transposed operand). base = element (row0, k0); ld = row
// stride in elements. Per ISA 7.12.2: lane L holds row L%16; elements 0..7
// are k = 8*(L/16)+e, elements 8..15 are k = 16 + 8*(L/16) + (e-8).
// Works for global (global_load_b128) and LDS (ds_load_b128) sources.
__device__ __forceinline__ v16bf load_frag(const bf16* __restrict__ base, int ld) {
    int lane = threadIdx.x & 31;
    int row  = lane & 15;
    int kh   = (lane >> 4) * 8;
    const bf16* p = base + (size_t)row * ld + kh;
    bf16x8 lo = *(const bf16x8*)(p);
    bf16x8 hi = *(const bf16x8*)(p + 16);
    v16bf f;
#pragma unroll
    for (int i = 0; i < 8; ++i) { f[i] = lo[i]; f[i + 8] = hi[i]; }
    return f;
}

// 16-byte global -> LDS copy, async on CDNA5 when the builtin exists.
__device__ __forceinline__ void copy16_g2l(const bf16* gsrc, bf16* ldst) {
#if HAVE_ASYNC_LDS
    __builtin_amdgcn_global_load_async_to_lds_b128(
        (as1_vi4*)gsrc, (as3_vi4*)ldst, 0, 0);
#else
    *(bf16x8*)ldst = *(const bf16x8*)gsrc;
#endif
}

// ---------------------------------------------------------------------------
// fp32 -> bf16 elementwise convert
__global__ void __launch_bounds__(256)
cvt_bf16_kernel(const float* __restrict__ in, bf16* __restrict__ out, int n) {
    int i = blockIdx.x * 256 + threadIdx.x;
    if (i < n) out[i] = (bf16)in[i];
}

// in [K,N] fp32 -> out [N,K] bf16 (so GEMM B-fragments are K-contiguous)
__global__ void __launch_bounds__(256)
transpose_cvt_kernel(const float* __restrict__ in, bf16* __restrict__ out,
                     int K, int N) {
    int i = blockIdx.x * 256 + threadIdx.x;
    if (i < K * N) {
        int n = i / K, k = i - n * K;
        out[i] = (bf16)in[(size_t)k * N + n];
    }
}

// ---------------------------------------------------------------------------
// C[M,N] f32 = A[M,K] bf16 @ BT[N,K] bf16  + bias[N]
// one wave computes a 32x64 tile: 2 A-frags x 4 B-frags -> 8 wmma per K step
__global__ void __launch_bounds__(256)
gemm_bf16_wmma(const bf16* __restrict__ A, const bf16* __restrict__ BT,
               const float* __restrict__ bias, float* __restrict__ C,
               int M, int N, int K) {
    int wave = (blockIdx.x * 256 + threadIdx.x) >> 5;
    int ntiles = N >> 6;                 // 64-wide N tiles
    int mt = wave / ntiles;
    int nt = wave - mt * ntiles;
    int m0 = mt * 32, n0 = nt * 64;
    if (m0 >= M) return;

    v8f acc[2][4] = {};
    for (int k0 = 0; k0 < K; k0 += 32) {
        if (k0 + 128 < K) {              // cover L2 latency for next steps
            __builtin_prefetch(A + (size_t)m0 * K + k0 + 128, 0, 0);
            __builtin_prefetch(BT + (size_t)n0 * K + k0 + 128, 0, 0);
        }
        v16bf a0 = load_frag(A + (size_t)m0 * K + k0, K);
        v16bf a1 = load_frag(A + (size_t)(m0 + 16) * K + k0, K);
        v16bf b[4];
#pragma unroll
        for (int j = 0; j < 4; ++j)
            b[j] = load_frag(BT + (size_t)(n0 + 16 * j) * K + k0, K);
#pragma unroll
        for (int j = 0; j < 4; ++j) {
            acc[0][j] = wmma_bf16(a0, b[j], acc[0][j]);
            acc[1][j] = wmma_bf16(a1, b[j], acc[1][j]);
        }
    }

    int lane = threadIdx.x & 31;
    int nl = lane & 15, half = lane >> 4;
#pragma unroll
    for (int i = 0; i < 2; ++i)
#pragma unroll
        for (int j = 0; j < 4; ++j) {
            int n = n0 + 16 * j + nl;
            float bv = bias[n];
#pragma unroll
            for (int r = 0; r < 8; ++r) {
                int m = m0 + 16 * i + r + 8 * half;
                C[(size_t)m * N + n] = acc[i][j][r] + bv;
            }
        }
}

// ---------------------------------------------------------------------------
// RoPE + split: fused [ROWS, 1536] f32 ->
//   Q  bf16 [B, NQH, S, 64]   (row-major, RoPE applied)
//   K  bf16 [B, NG,  S, 64]   (row-major, RoPE applied)
//   Vt bf16 [B, NG, 64, S ]   (transposed for PV A-fragments)
__global__ void __launch_bounds__(256)
rope_split_kernel(const float* __restrict__ fused, bf16* __restrict__ Q,
                  bf16* __restrict__ Kb, bf16* __restrict__ Vt) {
    int row = blockIdx.x;                 // b*SEQ + s
    int b = row / SEQ, s = row - b * SEQ;
    const float* fr = fused + (size_t)row * QKV_OUT;
    for (int c = threadIdx.x; c < QKV_OUT; c += 256) {
        float val = fr[c];
        if (c < MODEL_DIM) {                         // Q
            int h = c >> 6, hd = c & 63;
            float v2 = fr[hd < 32 ? c + 32 : c - 32];
            float rot = hd < 32 ? -v2 : v2;
            int j = hd & 31;
            float ang = (float)s * __powf(50000.0f, -(float)j * (1.0f / 32.0f));
            float sn, cs; __sincosf(ang, &sn, &cs);
            Q[(((size_t)(b * NQH + h)) * SEQ + s) * HD + hd] = (bf16)(val * cs + rot * sn);
        } else if (c < MODEL_DIM + 256) {            // K
            int cc = c - MODEL_DIM;
            int g = cc >> 6, hd = cc & 63;
            float v2 = fr[hd < 32 ? c + 32 : c - 32];
            float rot = hd < 32 ? -v2 : v2;
            int j = hd & 31;
            float ang = (float)s * __powf(50000.0f, -(float)j * (1.0f / 32.0f));
            float sn, cs; __sincosf(ang, &sn, &cs);
            Kb[(((size_t)(b * NG + g)) * SEQ + s) * HD + hd] = (bf16)(val * cs + rot * sn);
        } else {                                     // V (store transposed)
            int cc = c - (MODEL_DIM + 256);
            int g = cc >> 6, hd = cc & 63;
            Vt[(((size_t)(b * NG + g)) * HD + hd) * SEQ + s] = (bf16)val;
        }
    }
}

// ---------------------------------------------------------------------------
// Causal flash attention. One 256-thread block owns a 128-query macro-tile of
// one head; its 8 waves each run a 16-query WMMA tile and *share* the K / V^T
// chunk tiles, which are double-buffer staged into LDS with async-to-LDS
// copies (ASYNCcnt + barrier). Scores are computed transposed (S^T = K @ Q^T)
// so softmax rows live in lanes, and the P tile in C-layout is bit-for-bit
// the B-fragment of the PV wmma (O^T = V^T @ P) -> no cross-lane shuffles.
__global__ void __launch_bounds__(256)
flash_attn_kernel(const bf16* __restrict__ Q, const bf16* __restrict__ Kb,
                  const bf16* __restrict__ Vt, bf16* __restrict__ O) {
    __shared__ alignas(16) bf16 sK[2][32 * KPAD];   // 9216 B
    __shared__ alignas(16) bf16 sV[2][64 * VPAD];   // 10240 B

    const int nmacro = SEQ / MQ;          // 16
    int macro = blockIdx.x % nmacro;
    int bh    = blockIdx.x / nmacro;      // 0..31
    int b = bh >> 4, h = bh & 15;
    int g = h >> 2;
    int qbase = macro * MQ;

    const bf16* Qp = Q  + ((size_t)(b * NQH + h)) * SEQ * HD;
    const bf16* Kp = Kb + ((size_t)(b * NG  + g)) * SEQ * HD;
    const bf16* Vp = Vt + ((size_t)(b * NG  + g)) * HD * SEQ;

    int wave = threadIdx.x >> 5;
    int lane = threadIdx.x & 31;
    int nl = lane & 15, half = lane >> 4;
    int q0 = qbase + wave * 16;
    int qi = q0 + nl;                     // this lane's query index

    // Stage one 32-key chunk (K: 32x64, V^T: 64x32) into LDS buffer `buf`.
    auto stage = [&](int j0, int buf) {
        int t = threadIdx.x;
        {   // K tile: 256 x 16B
            int key = t >> 3, seg = t & 7;
            copy16_g2l(Kp + (size_t)(j0 + key) * HD + seg * 8,
                       &sK[buf][key * KPAD + seg * 8]);
        }
        {   // V^T tile: 256 x 16B
            int hd = t >> 2, seg = t & 3;
            copy16_g2l(Vp + (size_t)hd * SEQ + j0 + seg * 8,
                       &sV[buf][hd * VPAD + seg * 8]);
        }
    };

    // Q^T B-fragments (fixed for the whole wave): hd 0..31 and 32..63
    v16bf qf0 = load_frag(Qp + (size_t)q0 * HD + 0,  HD);
    v16bf qf1 = load_frag(Qp + (size_t)q0 * HD + 32, HD);

    v8f acc[4] = {};                      // O^T: 4 hd tiles x (lane=query)
    float mrow = -3.0e38f, lrow = 0.0f;
    const float scale = 0.125f;           // 1/sqrt(64)

    const int nchunks = (qbase + MQ) >> 5;
    stage(0, 0);
    for (int c = 0; c < nchunks; ++c) {
        int j0 = c * 32;
        if (c + 1 < nchunks) {
            stage(j0 + 32, (c + 1) & 1);
#if HAVE_ASYNC_LDS
            __builtin_amdgcn_s_wait_asynccnt(2);   // chunk c's copies done
#endif
        } else {
#if HAVE_ASYNC_LDS
            __builtin_amdgcn_s_wait_asynccnt(0);
#endif
        }
        __syncthreads();                  // all lanes' staging visible

        if (j0 < q0 + 16) {               // causal: this wave needs chunk c
            const bf16* kb = &sK[c & 1][0];
            const bf16* vb = &sV[c & 1][0];
            // ---- scores^T: two 16(key)x16(query) tiles over hd=64
            v8f sc0 = {}, sc1 = {};
            {
                v16bf ka0 = load_frag(kb + 0 * KPAD + 0,  KPAD);
                v16bf ka1 = load_frag(kb + 0 * KPAD + 32, KPAD);
                sc0 = wmma_bf16(ka0, qf0, sc0);
                sc0 = wmma_bf16(ka1, qf1, sc0);
                v16bf kb0 = load_frag(kb + 16 * KPAD + 0,  KPAD);
                v16bf kb1 = load_frag(kb + 16 * KPAD + 32, KPAD);
                sc1 = wmma_bf16(kb0, qf0, sc1);
                sc1 = wmma_bf16(kb1, qf1, sc1);
            }
            // ---- scale + causal mask + online softmax
            float p0[8], p1[8];
            float cmax = -3.0e38f;
#pragma unroll
            for (int r = 0; r < 8; ++r) {
                int key0 = j0 + r + 8 * half;
                int key1 = key0 + 16;
                float s0 = (key0 <= qi) ? sc0[r] * scale : -3.0e38f;
                float s1 = (key1 <= qi) ? sc1[r] * scale : -3.0e38f;
                p0[r] = s0; p1[r] = s1;
                cmax = fmaxf(cmax, fmaxf(s0, s1));
            }
            cmax = fmaxf(cmax, __shfl_xor(cmax, 16, 32));   // join key halves
            float mnew  = fmaxf(mrow, cmax);
            float alpha = __expf(mrow - mnew);
            float rsum = 0.0f;
            v16bf pf;                                       // P as B-fragment
#pragma unroll
            for (int r = 0; r < 8; ++r) {
                float e0 = __expf(p0[r] - mnew);
                float e1 = __expf(p1[r] - mnew);
                rsum += e0 + e1;
                pf[r]     = (bf16)e0;                       // keys j0..   (k 0..7)
                pf[r + 8] = (bf16)e1;                       // keys j0+16..(k 16..23)
            }
            rsum += __shfl_xor(rsum, 16, 32);
            lrow = lrow * alpha + rsum;
            mrow = mnew;
#pragma unroll
            for (int t = 0; t < 4; ++t)
#pragma unroll
                for (int r = 0; r < 8; ++r) acc[t][r] *= alpha;
            // ---- O^T += V^T(hd tile, 32 keys) @ P
#pragma unroll
            for (int t = 0; t < 4; ++t) {
                v16bf vf = load_frag(vb + (size_t)(16 * t) * VPAD, VPAD);
                acc[t] = wmma_bf16(vf, pf, acc[t]);
            }
        }
        __syncthreads();                  // buffer reuse fence
    }

    // ---- normalize, write attn output bf16 [ROWS, MODEL_DIM]
    float inv = 1.0f / lrow;
    size_t orow = ((size_t)(b * SEQ + qi)) * MODEL_DIM + (size_t)h * HD;
#pragma unroll
    for (int t = 0; t < 4; ++t) {
        bf16x8 ov;
#pragma unroll
        for (int r = 0; r < 8; ++r) ov[r] = (bf16)(acc[t][r] * inv);
        *(bf16x8*)(O + orow + 16 * t + 8 * half) = ov;   // 8 contiguous hd
    }
}

// ---------------------------------------------------------------------------
extern "C" void kernel_launch(void* const* d_in, const int* in_sizes, int n_in,
                              void* d_out, int out_size, void* d_ws, size_t ws_size,
                              hipStream_t stream) {
    const float* x     = (const float*)d_in[0];   // [2,2048,1024]
    const float* W_qkv = (const float*)d_in[1];   // [1024,1536]
    const float* b_qkv = (const float*)d_in[2];   // [1536]
    const float* W_out = (const float*)d_in[3];   // [1024,1024]
    const float* b_out = (const float*)d_in[4];   // [1024]
    float* out = (float*)d_out;                   // [2,2048,1024]

    // workspace carve-up (all region sizes are multiples of 256B)
    char* p = (char*)d_ws;
    bf16*  Xb    = (bf16*)p;  p += (size_t)ROWS * MODEL_DIM * 2;        // 8 MB
    bf16*  WqkvT = (bf16*)p;  p += (size_t)QKV_OUT * MODEL_DIM * 2;     // 3 MB
    bf16*  WoutT = (bf16*)p;  p += (size_t)MODEL_DIM * MODEL_DIM * 2;   // 2 MB
    float* fused = (float*)p; p += (size_t)ROWS * QKV_OUT * 4;          // 24 MB
    bf16*  Qb    = (bf16*)p;  p += (size_t)BATCH * NQH * SEQ * HD * 2;  // 8 MB
    bf16*  Kbuf  = (bf16*)p;  p += (size_t)BATCH * NG  * SEQ * HD * 2;  // 2 MB
    bf16*  Vt    = (bf16*)p;  p += (size_t)BATCH * NG  * HD * SEQ * 2;  // 2 MB
    bf16*  attnb = (bf16*)p;  p += (size_t)ROWS * MODEL_DIM * 2;        // 8 MB

    // 1) convert / transpose operands to bf16
    {
        int n = ROWS * MODEL_DIM;
        cvt_bf16_kernel<<<(n + 255) / 256, 256, 0, stream>>>(x, Xb, n);
        int nw = MODEL_DIM * QKV_OUT;
        transpose_cvt_kernel<<<(nw + 255) / 256, 256, 0, stream>>>(W_qkv, WqkvT, MODEL_DIM, QKV_OUT);
        int no = MODEL_DIM * MODEL_DIM;
        transpose_cvt_kernel<<<(no + 255) / 256, 256, 0, stream>>>(W_out, WoutT, MODEL_DIM, MODEL_DIM);
    }

    // 2) QKV projection: fused = Xb @ W_qkv + b_qkv   [4096 x 1536]
    {
        int waves = (ROWS / 32) * (QKV_OUT / 64);     // 3072
        gemm_bf16_wmma<<<waves / 8, 256, 0, stream>>>(Xb, WqkvT, b_qkv, fused,
                                                      ROWS, QKV_OUT, MODEL_DIM);
    }

    // 3) RoPE + head split (+ V transpose)
    rope_split_kernel<<<ROWS, 256, 0, stream>>>(fused, Qb, Kbuf, Vt);

    // 4) causal flash attention -> attnb bf16 [4096 x 1024]
    {
        int blocks = BATCH * NQH * (SEQ / MQ);        // 512
        flash_attn_kernel<<<blocks, 256, 0, stream>>>(Qb, Kbuf, Vt, attnb);
    }

    // 5) output projection: out = attnb @ W_out + b_out  [4096 x 1024] f32
    {
        int waves = (ROWS / 32) * (MODEL_DIM / 64);   // 2048
        gemm_bf16_wmma<<<waves / 8, 256, 0, stream>>>(attnb, WoutT, b_out, out,
                                                      ROWS, MODEL_DIM, MODEL_DIM);
    }
}
